// SlidingWindowAttention_81235011437107
// MI455X (gfx1250) — matrix operand order repaired and checked by reading, not compiled, not run on hardware
//
#include <hip/hip_runtime.h>
#include <hip/hip_bf16.h>

#define S_LEN  4096
#define D_DIM  1024
#define BATCH  4
#define HALF_W 64
#define QT_STR 1032   // padded LDS stride for Q tile (bank-conflict-free frags)

typedef __bf16 bf16_t;
typedef bf16_t v16bf __attribute__((ext_vector_type(16)));
typedef bf16_t v8bf  __attribute__((ext_vector_type(8)));
typedef float  v8f   __attribute__((ext_vector_type(8)));
typedef int    v4i   __attribute__((ext_vector_type(4)));

// Pointer-to-AS typedefs (attribute applies to the pointee).
typedef __attribute__((address_space(1))) v4i* gv4i_p;
typedef __attribute__((address_space(3))) v4i* lv4i_p;

#if __has_builtin(__builtin_amdgcn_global_load_async_to_lds_b128)
#define HAVE_ASYNC_LDS 1
#else
#define HAVE_ASYNC_LDS 0
#endif

__device__ __forceinline__ v16bf cat8(v8bf lo, v8bf hi) {
  return __builtin_shufflevector(lo, hi, 0,1,2,3,4,5,6,7,8,9,10,11,12,13,14,15);
}

// ---- Fragment loaders -------------------------------------------------------
// A-matrix 16x32 bf16: lanes 0-15: M=lane, K=0..15;
// lanes 16-31: M=lane-16, K=16..31.
__device__ __forceinline__ v16bf load_fragA_bf16(const bf16_t* base, int stride) {
  const int lane = threadIdx.x & 31;
  const bf16_t* p = base + (size_t)(lane & 15) * stride + ((lane >> 4) * 16);
  v8bf lo = *(const v8bf*)p;         // K = kb+0..7
  v8bf hi = *(const v8bf*)(p + 8);   // K = kb+8..15
  return cat8(lo, hi);
}
// B-matrix 32x16 bf16: lane -> N; lanes 0-15 hold K=0..15, lanes 16-31 K=16..31.
// Row-major storage indexed [n][k].
__device__ __forceinline__ v16bf load_fragB_bf16(const bf16_t* base, size_t stride,
                                                 int row, int k) {
  const int lane = threadIdx.x & 31;
  const bf16_t* p = base + (size_t)row * stride + k + ((lane >> 4) * 16);
  v8bf lo = *(const v8bf*)p;
  v8bf hi = *(const v8bf*)(p + 8);
  return cat8(lo, hi);
}
// B-fragment from transposed V [D][S], pairwise clamp of key index
// (out-of-range keys multiply P==0; we only need a safe address).
__device__ __forceinline__ v16bf load_fragB_vbt(const bf16_t* rowbase, int s0) {
  const int lane = threadIdx.x & 31;
  const int kb = (lane >> 4) * 16;
  union { unsigned u[8]; v16bf v; } f;
#pragma unroll
  for (int j = 0; j < 8; ++j) {
    int s = s0 + kb + 2 * j;
    s = s < 0 ? 0 : (s > S_LEN - 2 ? S_LEN - 2 : s);
    f.u[j] = *(const unsigned*)(rowbase + s);
  }
  return f.v;
}

// ---- Kernels ----------------------------------------------------------------
__global__ void zero_f4(float4* __restrict__ p, size_t n) {
  size_t i = (size_t)blockIdx.x * blockDim.x + threadIdx.x;
  size_t stride = (size_t)gridDim.x * blockDim.x;
  float4 z; z.x = z.y = z.z = z.w = 0.f;
  for (; i < n; i += stride) p[i] = z;
}

__global__ void cvt_bf16(const float* __restrict__ in, bf16_t* __restrict__ outp,
                         size_t n) {
  size_t i = ((size_t)blockIdx.x * blockDim.x + threadIdx.x) * 8;
  size_t stride = (size_t)gridDim.x * blockDim.x * 8;
  for (; i < n; i += stride) {
    float4 a = *(const float4*)(in + i);
    float4 b = *(const float4*)(in + i + 4);
    bf16_t t[8] = {(bf16_t)a.x, (bf16_t)a.y, (bf16_t)a.z, (bf16_t)a.w,
                   (bf16_t)b.x, (bf16_t)b.y, (bf16_t)b.z, (bf16_t)b.w};
    *(v8bf*)(outp + i) = *(v8bf*)t;
  }
}

// out[m][n] = sum_k x[m][k] * W[n][k] + bias[n], bf16 in, fp32 accum, bf16 out.
// transpose==0: out is [B*S][D]; transpose==1: out is [B][D][S] (for V).
// Block: 256 thr = 8 waves; wave tile 16(M) x 128(N); block tile 128 x 128.
__global__ __launch_bounds__(256) void proj_kernel(
    const bf16_t* __restrict__ x, const bf16_t* __restrict__ W,
    const float* __restrict__ bias, bf16_t* __restrict__ outp, int transpose) {
  const int lane = threadIdx.x & 31;
  const int wave = threadIdx.x >> 5;
  const int m0 = blockIdx.x * 128 + wave * 16;
  const int n0 = blockIdx.y * 128;

  const bf16_t* xrow = x + (size_t)m0 * D_DIM;
  v8f acc[8] = {};
  for (int k = 0; k < D_DIM; k += 32) {
    __builtin_prefetch(xrow + (size_t)(lane & 15) * D_DIM + k + 256, 0, 1);
    v16bf a = load_fragA_bf16(xrow + k, D_DIM);
#pragma unroll
    for (int t = 0; t < 8; ++t) {
      v16bf bm = load_fragB_bf16(W, D_DIM, n0 + t * 16 + (lane & 15), k);
      acc[t] = __builtin_amdgcn_wmma_f32_16x16x32_bf16(
          false, a, false, bm, (short)0, acc[t], false, false);
    }
  }

  const int col = lane & 15;
  const int rb = (lane >> 4) * 8;  // C layout: lane half -> M rows 0-7 / 8-15
  if (!transpose) {
#pragma unroll
    for (int t = 0; t < 8; ++t) {
      int n = n0 + t * 16 + col;
      float bv = bias[n];
#pragma unroll
      for (int j = 0; j < 8; ++j)
        outp[(size_t)(m0 + rb + j) * D_DIM + n] = (bf16_t)(acc[t][j] + bv);
    }
  } else {
    int b = m0 / S_LEN;
    int s = m0 % S_LEN;
#pragma unroll
    for (int t = 0; t < 8; ++t) {
      int n = n0 + t * 16 + col;
      float bv = bias[n];
      bf16_t tmp[8];
#pragma unroll
      for (int j = 0; j < 8; ++j) tmp[j] = (bf16_t)(acc[t][j] + bv);
      *(v8bf*)(outp + ((size_t)b * D_DIM + n) * S_LEN + s + rb) = *(v8bf*)tmp;
    }
  }
}

__global__ __launch_bounds__(256) void attn_kernel(
    const bf16_t* __restrict__ Qb, const bf16_t* __restrict__ Kb,
    const bf16_t* __restrict__ VbT, float* __restrict__ outp,
    float* __restrict__ attn) {
  __shared__ bf16_t qt[16][QT_STR];  // Q tile staged in LDS (padded stride)
  __shared__ float  sc[16][160];     // raw scores for this q-tile
  __shared__ bf16_t pt[16][160];     // softmax probs bf16, zero-padded to 160

  const int lane = threadIdx.x & 31;
  const int wave = threadIdx.x >> 5;
  const int b  = blockIdx.x >> 8;          // S/16 == 256 q-tiles per batch
  const int q0 = (blockIdx.x & 255) << 4;
  const int k0base = q0 - HALF_W;          // key index of stored column 0

  // ---- Stage Q tile (16 x 1024 bf16 = 32 KB) into LDS (async path on CDNA5)
  {
    const int row = threadIdx.x >> 4;         // 16 rows
    const int c0  = (threadIdx.x & 15) * 64;  // 16 chunks of 64 elems (128 B)
    const bf16_t* src = Qb + ((size_t)b * S_LEN + q0 + row) * D_DIM + c0;
#if HAVE_ASYNC_LDS
    // AS1 pointer value == flat address on AMD; integer round-trip drops const.
    gv4i_p g = (gv4i_p)(size_t)(const void*)src;
    lv4i_p l = (lv4i_p)(void*)&qt[row][c0];
#pragma unroll
    for (int i = 0; i < 8; ++i)
      __builtin_amdgcn_global_load_async_to_lds_b128(g + i, l + i, 0, 0);
#if __has_builtin(__builtin_amdgcn_s_wait_asynccnt)
    __builtin_amdgcn_s_wait_asynccnt(0);
#else
    asm volatile("s_wait_asynccnt 0x0" ::: "memory");
#endif
#else
#pragma unroll
    for (int i = 0; i < 8; ++i)
      *(v8bf*)&qt[row][c0 + i * 8] = *(const v8bf*)(src + i * 8);
#endif
  }
  __syncthreads();

  // ---- Phase A: scores = Q . K^T over 9 key tiles (144 keys covers the band)
  const bf16_t* Kbbase = Kb + (size_t)b * S_LEN * D_DIM;
  for (int kt = wave; kt < 9; kt += 8) {
    int krow = k0base + kt * 16 + (lane & 15);
    krow = krow < 0 ? 0 : (krow >= S_LEN ? S_LEN - 1 : krow);  // masked later
    v8f acc = {};
    for (int k = 0; k < D_DIM; k += 32) {
      v16bf a  = load_fragA_bf16(&qt[0][0] + k, QT_STR);
      v16bf bm = load_fragB_bf16(Kbbase, (size_t)D_DIM, krow, k);
      acc = __builtin_amdgcn_wmma_f32_16x16x32_bf16(
          false, a, false, bm, (short)0, acc, false, false);
    }
    const int col = kt * 16 + (lane & 15);
    const int rb = (lane >> 4) * 8;
#pragma unroll
    for (int j = 0; j < 8; ++j) sc[rb + j][col] = acc[j];
  }
  __syncthreads();

  // ---- Phase B: banded softmax, 2 rows per wave; write attn band fp32
#pragma unroll
  for (int rr = 0; rr < 2; ++rr) {
    const int r = wave * 2 + rr;
    const int qi = q0 + r;
    float m = -3.0e38f;
#pragma unroll
    for (int t = 0; t < 5; ++t) {
      int j = lane + 32 * t;
      int jg = k0base + j;
      bool valid = (j >= r) && (j <= r + 2 * HALF_W) && (jg >= 0) && (jg < S_LEN);
      float v = valid ? sc[r][j] * 0.03125f : -3.0e38f;  // 1/sqrt(D) = 1/32
      m = fmaxf(m, v);
    }
#pragma unroll
    for (int o = 16; o > 0; o >>= 1) m = fmaxf(m, __shfl_xor(m, o, 32));
    float sum = 0.f;
    float pv[5];
#pragma unroll
    for (int t = 0; t < 5; ++t) {
      int j = lane + 32 * t;
      int jg = k0base + j;
      bool valid = (j >= r) && (j <= r + 2 * HALF_W) && (jg >= 0) && (jg < S_LEN);
      float p = valid ? __expf(sc[r][j] * 0.03125f - m) : 0.f;
      pv[t] = p;
      sum += p;
    }
#pragma unroll
    for (int o = 16; o > 0; o >>= 1) sum += __shfl_xor(sum, o, 32);
    float inv = 1.f / sum;
#pragma unroll
    for (int t = 0; t < 5; ++t) {
      int j = lane + 32 * t;
      int jg = k0base + j;
      float p = pv[t] * inv;
      pt[r][j] = (bf16_t)p;  // zero outside band and in the 144..159 pad
      bool valid = (j >= r) && (j <= r + 2 * HALF_W) && (jg >= 0) && (jg < S_LEN);
      if (valid) attn[((size_t)b * S_LEN + qi) * S_LEN + jg] = p;
    }
  }
  __syncthreads();

  // ---- Phase C: out = P . V  (each wave: 8 n-tiles of 16; 5 K-chunks of 32)
  const bf16_t* vb = VbT + (size_t)b * D_DIM * S_LEN;
  for (int nt = wave * 8; nt < wave * 8 + 8; ++nt) {
    const int n0 = nt * 16;
    const bf16_t* rowbase = vb + (size_t)(n0 + (lane & 15)) * S_LEN;
    v8f acc = {};
#pragma unroll
    for (int c = 0; c < 5; ++c) {
      v16bf a  = load_fragA_bf16(&pt[0][0] + c * 32, 160);
      v16bf bm = load_fragB_vbt(rowbase, k0base + c * 32);
      acc = __builtin_amdgcn_wmma_f32_16x16x32_bf16(
          false, a, false, bm, (short)0, acc, false, false);
    }
    const int col = n0 + (lane & 15);
    const int rb = (lane >> 4) * 8;
#pragma unroll
    for (int j = 0; j < 8; ++j)
      outp[((size_t)b * S_LEN + q0 + rb + j) * D_DIM + col] = acc[j];
  }
}

// ---- Host launch ------------------------------------------------------------
extern "C" void kernel_launch(void* const* d_in, const int* in_sizes, int n_in,
                              void* d_out, int out_size, void* d_ws, size_t ws_size,
                              hipStream_t stream) {
  const float* x  = (const float*)d_in[0];
  const float* Wq = (const float*)d_in[1];
  const float* bq = (const float*)d_in[2];
  const float* Wk = (const float*)d_in[3];
  const float* bk = (const float*)d_in[4];
  const float* Wv = (const float*)d_in[5];
  const float* bv = (const float*)d_in[6];

  float* out  = (float*)d_out;
  float* attn = out + (size_t)BATCH * S_LEN * D_DIM;

  const size_t bsd = (size_t)BATCH * S_LEN * D_DIM;
  const size_t dd  = (size_t)D_DIM * D_DIM;
  bf16_t* Qb  = (bf16_t*)d_ws;   // bf16 [B*S][D]
  bf16_t* Kb  = Qb + bsd;        // bf16 [B*S][D]
  bf16_t* VbT = Kb + bsd;        // bf16 [B][D][S] (transposed for PV B-frags)
  bf16_t* xb  = VbT + bsd;       // bf16 [B*S][D]  (pre-converted activations)
  bf16_t* Wqb = xb + bsd;        // bf16 [D][D]
  bf16_t* Wkb = Wqb + dd;
  bf16_t* Wvb = Wkb + dd;

  // attn_weights must be exactly 0 outside the band (buffer is poisoned)
  size_t n4 = (size_t)BATCH * S_LEN * S_LEN / 4;
  zero_f4<<<4096, 256, 0, stream>>>((float4*)attn, n4);

  // one-shot fp32 -> bf16 conversion (bandwidth-bound, removes inner-loop cvt)
  cvt_bf16<<<4096, 256, 0, stream>>>(x, xb, bsd);
  cvt_bf16<<<512, 256, 0, stream>>>(Wq, Wqb, dd);
  cvt_bf16<<<512, 256, 0, stream>>>(Wk, Wkb, dd);
  cvt_bf16<<<512, 256, 0, stream>>>(Wv, Wvb, dd);

  dim3 pgrid((BATCH * S_LEN) / 128, D_DIM / 128);
  proj_kernel<<<pgrid, 256, 0, stream>>>(xb, Wqb, bq, Qb, 0);
  proj_kernel<<<pgrid, 256, 0, stream>>>(xb, Wkb, bk, Kb, 0);
  proj_kernel<<<pgrid, 256, 0, stream>>>(xb, Wvb, bv, VbT, 1);

  attn_kernel<<<BATCH * (S_LEN / 16), 256, 0, stream>>>(Qb, Kb, VbT, out, attn);
}
